// LandslideEEGMoE_23012434772545
// MI455X (gfx1250) — compile-verified, matrix operand order checked by reading, not run on hardware
//
#include <hip/hip_runtime.h>
#include <hip/hip_bf16.h>
#include <math.h>

// ---------------------------------------------------------------------------
// Model constants
// ---------------------------------------------------------------------------
#define B_    16
#define C_    5
#define Him_  256
#define Wim_  256
#define Ppat_ 8
#define SF_   64
#define ED_   128
#define HS_   512
#define NH_   4
#define HD_   128     // HS/NH
#define FF_   2048
#define L_    4
#define NE_   6
#define NS_   2
#define NC_   2
#define S_TOK 1024    // (256/8)^2
#define T_TOK (B_*S_TOK)   // 16384 tokens

typedef __bf16 bf16;
typedef bf16  v16bf __attribute__((ext_vector_type(16)));
typedef bf16  v8bf  __attribute__((ext_vector_type(8)));
typedef float v8f   __attribute__((ext_vector_type(8)));

// ---------------------------------------------------------------------------
// CDNA5 primitives
// ---------------------------------------------------------------------------
__device__ __forceinline__ v8f wmma_bf16(v16bf a, v16bf b, v8f c) {
    return __builtin_amdgcn_wmma_f32_16x16x32_bf16(
        /*neg_a=*/false, a, /*neg_b=*/false, b,
        /*c_mod=*/(short)0, c, /*reuse_a=*/false, /*reuse_b=*/false);
}

// Async DMA: 16 bytes global -> LDS, tracked by ASYNCcnt (no VGPR staging).
__device__ __forceinline__ void async_b128(unsigned lds_off, const void* gptr) {
    asm volatile("global_load_async_to_lds_b128 %0, %1, off"
                 :: "v"(lds_off), "v"(gptr) : "memory");
}

__device__ __forceinline__ void wait_async0() {
#if __has_builtin(__builtin_amdgcn_s_wait_asynccnt)
    __builtin_amdgcn_s_wait_asynccnt(0);
#else
    asm volatile("s_wait_asynccnt 0x0" ::: "memory");
#endif
}

// Generic LDS pointer -> 32-bit LDS byte offset (low 32 bits of flat address).
__device__ __forceinline__ unsigned lds_off(const void* p) {
    return (unsigned)(size_t)p;
}

// Load a 16x32 (bf16) A/B fragment from a row-major tile [rows][stride].
// Contiguous k order (k = 16*(lane>>4) + s), same for A and B: lane&15 selects row, lane>>4 selects k half.
__device__ __forceinline__ v16bf ld_frag(const bf16* base, int row, int kbase,
                                         int stride, int lane) {
    const bf16* p = base + (row + (lane & 15)) * stride + kbase + ((lane >> 4) << 4);
    v8bf lo = *(const v8bf*)p;
    v8bf hi = *(const v8bf*)(p + 8);
    v16bf r;
#pragma unroll
    for (int i = 0; i < 8; ++i) { r[i] = lo[i]; r[i + 8] = hi[i]; }
    return r;
}

__device__ __forceinline__ void zero16(bf16* p) {
    v8bf z;
#pragma unroll
    for (int i = 0; i < 8; ++i) z[i] = (bf16)0.0f;
    *(v8bf*)p = z;
}

// ---------------------------------------------------------------------------
// Generic bf16 GEMM: C[M,N] = act(A[M,K] @ W[N,K]^T + bias) * rowscale, f32 acc
//   act: 0=none, 1=relu, 2=gelu(erf); rowscale -> MoE combine; accum: Cf += y
// 128x128x32 block tile, 256 threads = 8 waves (2x4), wave tile 64x32.
// Double-buffered LDS fed by global_load_async_to_lds_b128; fragment loads
// are batched ahead of the WMMA burst to amortize DScnt waits.
// ---------------------------------------------------------------------------
__global__ __launch_bounds__(256) void gemm_bf16_kernel(
    const bf16* __restrict__ A, const bf16* __restrict__ Bw,
    const float* __restrict__ bias,
    float* __restrict__ Cf, bf16* __restrict__ Cb,
    const float* __restrict__ rowscale, int rs_stride,
    int M, int N, int K, int act, int accum)
{
    __shared__ bf16 sA[2][128 * 48];
    __shared__ bf16 sB[2][128 * 48];

    const int tid  = threadIdx.x;
    const int lane = tid & 31;
    const int wave = tid >> 5;
    const int wm   = wave >> 2;    // 0..1
    const int wn   = wave & 3;     // 0..3
    const int bm0  = blockIdx.y * 128;
    const int bn0  = blockIdx.x * 128;

    v8f acc[4][2];
#pragma unroll
    for (int i = 0; i < 4; ++i)
#pragma unroll
        for (int j = 0; j < 2; ++j)
#pragma unroll
            for (int r = 0; r < 8; ++r) acc[i][j][r] = 0.0f;

    const int lr = tid >> 1;          // 0..127 : tile row this thread stages
    const int lc = (tid & 1) * 16;    // 0 or 16 : tile col base (16 elems)
    const bool arow = (bm0 + lr) < M;
    const bool brow = (bn0 + lr) < N;
    const bf16* ga = A  + (size_t)(bm0 + lr) * K + lc;
    const bf16* gb = Bw + (size_t)(bn0 + lr) * K + lc;

    const int nk = K >> 5;

    // stage tile k0 (elems) into buffer `buf`
    auto stage = [&](int buf, int k0) {
        bf16* da = &sA[buf][lr * 48 + lc];
        bf16* db = &sB[buf][lr * 48 + lc];
        if (arow) {
            async_b128(lds_off(da),     ga + k0);
            async_b128(lds_off(da + 8), ga + k0 + 8);
        } else { zero16(da); zero16(da + 8); }
        if (brow) {
            async_b128(lds_off(db),     gb + k0);
            async_b128(lds_off(db + 8), gb + k0 + 8);
        } else { zero16(db); zero16(db + 8); }
    };

    stage(0, 0);

    for (int kk = 0; kk < nk; ++kk) {
        const int buf = kk & 1;
        wait_async0();        // this wave's async writes are in LDS
        __syncthreads();      // everyone's writes visible; prior reads retired
        if (kk + 1 < nk) stage(buf ^ 1, (kk + 1) << 5);

        // batch all fragment loads, then burst the WMMAs
        v16bf bfrag[2], afrag[4];
#pragma unroll
        for (int j = 0; j < 2; ++j)
            bfrag[j] = ld_frag(sB[buf], wn * 32 + j * 16, 0, 48, lane);
#pragma unroll
        for (int i = 0; i < 4; ++i)
            afrag[i] = ld_frag(sA[buf], wm * 64 + i * 16, 0, 48, lane);
#pragma unroll
        for (int i = 0; i < 4; ++i)
#pragma unroll
            for (int j = 0; j < 2; ++j)
                acc[i][j] = wmma_bf16(afrag[i], bfrag[j], acc[i][j]);
    }

    // --- epilogue ---
    const int coln = lane & 15;
    const int half = lane >> 4;
#pragma unroll
    for (int i = 0; i < 4; ++i) {
#pragma unroll
        for (int j = 0; j < 2; ++j) {
#pragma unroll
            for (int r = 0; r < 8; ++r) {
                const int gm = bm0 + wm * 64 + i * 16 + r + half * 8;
                const int gn = bn0 + wn * 32 + j * 16 + coln;
                if (gm < M && gn < N) {
                    float y = acc[i][j][r];
                    if (bias) y += bias[gn];
                    if (act == 1)      y = fmaxf(y, 0.0f);
                    else if (act == 2) y = 0.5f * y * (1.0f + erff(y * 0.70710678118f));
                    if (rowscale) y *= rowscale[(size_t)gm * rs_stride];
                    const size_t o = (size_t)gm * N + gn;
                    if (Cf) Cf[o] = accum ? (Cf[o] + y) : y;
                    if (Cb) Cb[o] = (bf16)y;
                }
            }
        }
    }
}

// ---------------------------------------------------------------------------
// Flash attention (per b, head, 128-query tile). qkv: [T, 3*HS] bf16,
// out: [T, HS] bf16. 256 threads = 8 waves; each wave owns 16 q rows x d=128.
// K tile staged by async DMA; V staged transposed (manual); P via LDS.
// ---------------------------------------------------------------------------
__global__ __launch_bounds__(256) void attn_kernel(
    const bf16* __restrict__ qkv, bf16* __restrict__ out)
{
    __shared__ bf16 sK[64 * 144];    // K tile  [kv=64][d stride 144]
    __shared__ bf16 sVt[128 * 80];   // V^T     [d=128][kv stride 80]
    __shared__ bf16 sP[128 * 80];    // P tile  [q=128][kv stride 80]

    const int tid  = threadIdx.x;
    const int lane = tid & 31;
    const int wave = tid >> 5;
    const int qt   = blockIdx.x;          // 0..7
    const int h    = blockIdx.y;          // 0..3
    const int b    = blockIdx.z;          // 0..15
    const int q0   = qt * 128;
    const int RS   = 3 * HS_;             // 1536

    const bf16* qbase = qkv + (size_t)b * S_TOK * RS + h * HD_;
    const bf16* kbase = qbase + HS_;
    const bf16* vbase = qbase + 2 * HS_;

    // Q fragments live in registers: 4 chunks of 32 along d.
    v16bf qf[4];
    {
        const int qr = q0 + wave * 16 + (lane & 15);
        const int ko = (lane >> 4) << 4;
#pragma unroll
        for (int c = 0; c < 4; ++c) {
            const bf16* p = qbase + (size_t)qr * RS + c * 32 + ko;
            v8bf lo = *(const v8bf*)p;
            v8bf hi = *(const v8bf*)(p + 8);
#pragma unroll
            for (int i = 0; i < 8; ++i) { qf[c][i] = lo[i]; qf[c][i + 8] = hi[i]; }
        }
    }

    v8f o[8];
#pragma unroll
    for (int j = 0; j < 8; ++j)
#pragma unroll
        for (int r = 0; r < 8; ++r) o[j][r] = 0.0f;
    float m8[8], l8[8];
#pragma unroll
    for (int r = 0; r < 8; ++r) { m8[r] = -INFINITY; l8[r] = 0.0f; }

    const float scale = 0.08838834764831845f;   // 1/sqrt(128)
    const int wq = wave * 16;

    for (int kv0 = 0; kv0 < S_TOK; kv0 += 64) {
        // stage K tile via async DMA (row-major along d)
        {
            const int r  = tid >> 2;
            const int cb = (tid & 3) * 8;
#pragma unroll
            for (int it = 0; it < 4; ++it) {
                const int col = cb + it * 32;
                async_b128(lds_off(&sK[r * 144 + col]),
                           kbase + (size_t)(kv0 + r) * RS + col);
            }
        }
        // stage V transposed: sVt[d][kv]
        {
            const int kv = tid >> 2;
            const int db = (tid & 3) * 8;
#pragma unroll
            for (int it = 0; it < 4; ++it) {
                const int d = db + it * 32;
                v8bf v = *(const v8bf*)(vbase + (size_t)(kv0 + kv) * RS + d);
#pragma unroll
                for (int j = 0; j < 8; ++j) sVt[(d + j) * 80 + kv] = v[j];
            }
        }
        wait_async0();
        __syncthreads();

        // S tile (16 q x 64 kv) per wave; batch K-fragments per d-chunk
        v8f sacc[4];
#pragma unroll
        for (int j = 0; j < 4; ++j)
#pragma unroll
            for (int r = 0; r < 8; ++r) sacc[j][r] = 0.0f;
#pragma unroll
        for (int c = 0; c < 4; ++c) {
            v16bf kf[4];
#pragma unroll
            for (int j = 0; j < 4; ++j)
                kf[j] = ld_frag(sK, j * 16, c * 32, 144, lane);
#pragma unroll
            for (int j = 0; j < 4; ++j)
                sacc[j] = wmma_bf16(qf[c], kf[j], sacc[j]);
        }
#pragma unroll
        for (int j = 0; j < 4; ++j)
#pragma unroll
            for (int r = 0; r < 8; ++r) sacc[j][r] *= scale;

        // online softmax: row max / rescale
#pragma unroll
        for (int r = 0; r < 8; ++r) {
            float v = fmaxf(fmaxf(sacc[0][r], sacc[1][r]),
                            fmaxf(sacc[2][r], sacc[3][r]));
            v = fmaxf(v, __shfl_xor(v, 1, 16));
            v = fmaxf(v, __shfl_xor(v, 2, 16));
            v = fmaxf(v, __shfl_xor(v, 4, 16));
            v = fmaxf(v, __shfl_xor(v, 8, 16));
            const float mnew  = fmaxf(m8[r], v);
            const float alpha = __expf(m8[r] - mnew);
            m8[r] = mnew;
            l8[r] *= alpha;
#pragma unroll
            for (int j = 0; j < 8; ++j) o[j][r] *= alpha;
        }
        // P = exp(S - m), row sums, stash P tile (wave-private rows)
#pragma unroll
        for (int r = 0; r < 8; ++r) {
            const int prow = wq + r + (lane >> 4) * 8;
            float rs = 0.0f;
#pragma unroll
            for (int j = 0; j < 4; ++j) {
                float p = __expf(sacc[j][r] - m8[r]);
                rs += p;
                sP[prow * 80 + j * 16 + (lane & 15)] = (bf16)p;
            }
            rs += __shfl_xor(rs, 1, 16);
            rs += __shfl_xor(rs, 2, 16);
            rs += __shfl_xor(rs, 4, 16);
            rs += __shfl_xor(rs, 8, 16);
            l8[r] += rs;
        }
        // O += P @ V (batch V-fragments in groups of 4)
#pragma unroll
        for (int kc = 0; kc < 2; ++kc) {
            v16bf pf = ld_frag(sP, wq, kc * 32, 80, lane);
#pragma unroll
            for (int jg = 0; jg < 2; ++jg) {
                v16bf vf[4];
#pragma unroll
                for (int j = 0; j < 4; ++j)
                    vf[j] = ld_frag(sVt, (jg * 4 + j) * 16, kc * 32, 80, lane);
#pragma unroll
                for (int j = 0; j < 4; ++j)
                    o[jg * 4 + j] = wmma_bf16(pf, vf[j], o[jg * 4 + j]);
            }
        }
        __syncthreads();
    }

    // write normalized output
    {
        const int qg = q0 + wq;
#pragma unroll
        for (int j = 0; j < 8; ++j) {
#pragma unroll
            for (int r = 0; r < 8; ++r) {
                const int row = qg + r + (lane >> 4) * 8;
                const float val = o[j][r] / l8[r];
                out[((size_t)(b * S_TOK + row)) * HS_ + h * HD_ + j * 16 + (lane & 15)]
                    = (bf16)val;
            }
        }
    }
}

// ---------------------------------------------------------------------------
// LayerNorm over D=512: y = LN(xa [+ xb]) * g + b ; optional f32/bf16 outputs.
// ---------------------------------------------------------------------------
__global__ __launch_bounds__(128) void ln_kernel(
    const float* __restrict__ xa, const float* __restrict__ xb,
    const float* __restrict__ g, const float* __restrict__ be,
    float* __restrict__ of, bf16* __restrict__ ob)
{
    __shared__ float red[128];
    const int row = blockIdx.x;
    const int tid = threadIdx.x;
    float v[4];
    float s = 0.0f;
#pragma unroll
    for (int i = 0; i < 4; ++i) {
        const int c = tid + i * 128;
        float x = xa[(size_t)row * HS_ + c];
        if (xb) x += xb[(size_t)row * HS_ + c];
        v[i] = x; s += x;
    }
    red[tid] = s; __syncthreads();
    for (int st = 64; st > 0; st >>= 1) {
        if (tid < st) red[tid] += red[tid + st];
        __syncthreads();
    }
    const float mean = red[0] * (1.0f / HS_);
    __syncthreads();
    float s2 = 0.0f;
#pragma unroll
    for (int i = 0; i < 4; ++i) { const float d = v[i] - mean; s2 += d * d; }
    red[tid] = s2; __syncthreads();
    for (int st = 64; st > 0; st >>= 1) {
        if (tid < st) red[tid] += red[tid + st];
        __syncthreads();
    }
    const float inv = rsqrtf(red[0] * (1.0f / HS_) + 1e-5f);
#pragma unroll
    for (int i = 0; i < 4; ++i) {
        const int c = tid + i * 128;
        const float y = (v[i] - mean) * inv * g[c] + be[c];
        if (of) of[(size_t)row * HS_ + c] = y;
        if (ob) ob[(size_t)row * HS_ + c] = (bf16)y;
    }
}

// ---------------------------------------------------------------------------
// Routers
// ---------------------------------------------------------------------------
__global__ void spec_router_kernel(const float* __restrict__ logits,
                                   float* __restrict__ comb)
{
    const int t = blockIdx.x * blockDim.x + threadIdx.x;
    if (t >= T_TOK) return;
    float p[NE_];
    float m = -INFINITY;
#pragma unroll
    for (int e = 0; e < NE_; ++e) { p[e] = logits[t * NE_ + e]; m = fmaxf(m, p[e]); }
    float s = 0.0f;
#pragma unroll
    for (int e = 0; e < NE_; ++e) { p[e] = __expf(p[e] - m); s += p[e]; }
    const float inv = 1.0f / s;
#pragma unroll
    for (int e = 0; e < NE_; ++e) p[e] *= inv;
    int i0 = 0;
#pragma unroll
    for (int e = 1; e < NE_; ++e) if (p[e] > p[i0]) i0 = e;
    int i1 = (i0 == 0) ? 1 : 0;
#pragma unroll
    for (int e = 0; e < NE_; ++e) if (e != i0 && p[e] > p[i1]) i1 = e;
    const float tt = 1.0f / (p[i0] + p[i1] + 1e-9f);
#pragma unroll
    for (int e = 0; e < NE_; ++e)
        comb[t * NE_ + e] = (e == i0) ? p[i0] * tt : ((e == i1) ? p[i1] * tt : 0.0f);
}

__global__ void shr_router_kernel(const float* __restrict__ logits,
                                  float* __restrict__ sp)
{
    const int t = blockIdx.x * blockDim.x + threadIdx.x;
    if (t >= T_TOK) return;
    const float a = logits[t * NS_ + 0], b = logits[t * NS_ + 1];
    const float m = fmaxf(a, b);
    const float ea = __expf(a - m), eb = __expf(b - m);
    const float inv = 1.0f / (ea + eb);
    sp[t * NS_ + 0] = ea * inv;
    sp[t * NS_ + 1] = eb * inv;
}

// ---------------------------------------------------------------------------
// Misc small kernels
// ---------------------------------------------------------------------------
__global__ void cast_kernel(const float* __restrict__ in, bf16* __restrict__ out,
                            size_t n)
{
    size_t i = (size_t)blockIdx.x * blockDim.x + threadIdx.x;
    const size_t stride = (size_t)gridDim.x * blockDim.x;
    for (; i < n; i += stride) out[i] = (bf16)in[i];
}

__global__ void patchify_kernel(const float* __restrict__ x, bf16* __restrict__ t)
{
    size_t i = (size_t)blockIdx.x * blockDim.x + threadIdx.x;
    const size_t total = (size_t)T_TOK * (C_ * Ppat_ * Ppat_);
    if (i >= total) return;
    const int inner = (int)(i % 320);
    const size_t tok = i / 320;
    const int b = (int)(tok / S_TOK);
    const int s = (int)(tok % S_TOK);
    const int sh = s >> 5, sw = s & 31;
    const int c = inner >> 6;
    const int rr = inner & 63;
    const int py = rr >> 3, px = rr & 7;
    const size_t xi = (((size_t)(b * C_ + c) * Him_) + sh * 8 + py) * Wim_ + sw * 8 + px;
    t[i] = (bf16)x[xi];
}

__global__ void peadd_kernel(const float* __restrict__ z, bf16* __restrict__ zb)
{
    const int i = blockIdx.x * blockDim.x + threadIdx.x;
    if (i >= T_TOK * ED_) return;
    const int d = i & (ED_ - 1);
    const int s = (i / ED_) & (S_TOK - 1);
    const int k = d >> 1;
    const float div = __expf(-(float)(2 * k) * (9.210340371976184f / (float)ED_));
    const float ang = (float)s * div;
    const float pe = (d & 1) ? __cosf(ang) : __sinf(ang);
    zb[i] = (bf16)(z[i] + pe);
}

__global__ void pool_kernel(const float* __restrict__ h, bf16* __restrict__ pooled)
{
    const int idx = blockIdx.x * blockDim.x + threadIdx.x;
    if (idx >= B_ * HS_) return;
    const int b = idx / HS_, d = idx % HS_;
    float s = 0.0f;
    for (int ss = 0; ss < S_TOK; ++ss)
        s += h[((size_t)(b * S_TOK + ss)) * HS_ + d];
    pooled[idx] = (bf16)(s * (1.0f / S_TOK));
}

// ---------------------------------------------------------------------------
// Host launcher
// ---------------------------------------------------------------------------
static inline size_t align_up(size_t x) { return (x + 255) & ~(size_t)255; }

extern "C" void kernel_launch(void* const* d_in, const int* in_sizes, int n_in,
                              void* d_out, int out_size, void* d_ws, size_t ws_size,
                              hipStream_t stream)
{
    (void)in_sizes; (void)n_in; (void)out_size; (void)ws_size;

    const float* x          = (const float*)d_in[0];
    const float* tok_w1     = (const float*)d_in[1];
    const float* tok_b1     = (const float*)d_in[2];
    const float* tok_w2     = (const float*)d_in[3];
    const float* tok_b2     = (const float*)d_in[4];
    const float* proj_w     = (const float*)d_in[5];
    const float* proj_b     = (const float*)d_in[6];
    const float* attn_wqkv  = (const float*)d_in[7];
    const float* attn_bqkv  = (const float*)d_in[8];
    const float* attn_wo    = (const float*)d_in[9];
    const float* attn_bo    = (const float*)d_in[10];
    const float* ln1_g      = (const float*)d_in[11];
    const float* ln1_b      = (const float*)d_in[12];
    const float* spec_rw    = (const float*)d_in[13];
    const float* spec_f1w   = (const float*)d_in[14];
    const float* spec_f1b   = (const float*)d_in[15];
    const float* spec_f2w   = (const float*)d_in[16];
    const float* spec_f2b   = (const float*)d_in[17];
    const float* shr_rw     = (const float*)d_in[18];
    const float* shr_f1w    = (const float*)d_in[19];
    const float* shr_f1b    = (const float*)d_in[20];
    const float* shr_f2w    = (const float*)d_in[21];
    const float* shr_f2b    = (const float*)d_in[22];
    const float* lnm_g      = (const float*)d_in[23];
    const float* lnm_b      = (const float*)d_in[24];
    const float* ln2_g      = (const float*)d_in[25];
    const float* ln2_b      = (const float*)d_in[26];
    const float* cls_w      = (const float*)d_in[27];
    const float* cls_b      = (const float*)d_in[28];

    char* ws = (char*)d_ws;
    size_t off = 0;
    auto alloc = [&](size_t bytes) -> void* {
        void* p = ws + off; off = align_up(off + bytes); return p;
    };

    const size_t T = T_TOK;
    // bf16 weights
    bf16* w_tok1  = (bf16*)alloc((size_t)SF_ * 320 * 2);
    bf16* w_tok2  = (bf16*)alloc((size_t)ED_ * SF_ * 2);
    bf16* w_proj  = (bf16*)alloc((size_t)HS_ * ED_ * 2);
    bf16* w_qkv   = (bf16*)alloc((size_t)L_ * 3 * HS_ * HS_ * 2);
    bf16* w_wo    = (bf16*)alloc((size_t)L_ * HS_ * HS_ * 2);
    bf16* w_specr = (bf16*)alloc((size_t)L_ * NE_ * HS_ * 2);
    bf16* w_spec1 = (bf16*)alloc((size_t)L_ * NE_ * FF_ * HS_ * 2);
    bf16* w_spec2 = (bf16*)alloc((size_t)L_ * NE_ * HS_ * FF_ * 2);
    bf16* w_shrr  = (bf16*)alloc((size_t)L_ * NS_ * HS_ * 2);
    bf16* w_shr1  = (bf16*)alloc((size_t)L_ * NS_ * FF_ * HS_ * 2);
    bf16* w_shr2  = (bf16*)alloc((size_t)L_ * NS_ * HS_ * FF_ * 2);
    bf16* w_cls   = (bf16*)alloc((size_t)NC_ * HS_ * 2);
    // activations
    bf16*  t_bf   = (bf16*)alloc(T * 320 * 2);
    bf16*  z1_bf  = (bf16*)alloc(T * SF_ * 2);
    float* z2_f   = (float*)alloc(T * ED_ * 4);
    bf16*  z2_bf  = (bf16*)alloc(T * ED_ * 2);
    float* h_f    = (float*)alloc(T * HS_ * 4);
    bf16*  h_bf   = (bf16*)alloc(T * HS_ * 2);
    bf16*  qkv_bf = (bf16*)alloc(T * 3 * HS_ * 2);
    bf16*  ao_bf  = (bf16*)alloc(T * HS_ * 2);
    float* tmp_f  = (float*)alloc(T * HS_ * 4);
    float* m_f    = (float*)alloc(T * HS_ * 4);
    float* moe_f  = (float*)alloc(T * HS_ * 4);
    bf16*  hid_bf = (bf16*)alloc(T * FF_ * 2);
    float* lg1_f  = (float*)alloc(T * NE_ * 4);
    float* lg2_f  = (float*)alloc(T * NS_ * 4);
    float* comb_f = (float*)alloc(T * NE_ * 4);
    float* sp_f   = (float*)alloc(T * NS_ * 4);
    bf16*  pool_bf= (bf16*)alloc((size_t)B_ * HS_ * 2);

    auto cast = [&](const float* src, bf16* dst, size_t n) {
        int grid = (int)(((n + 255) / 256) > 8192 ? 8192 : ((n + 255) / 256));
        cast_kernel<<<grid, 256, 0, stream>>>(src, dst, n);
    };
    cast(tok_w1,   w_tok1,  (size_t)SF_ * 320);
    cast(tok_w2,   w_tok2,  (size_t)ED_ * SF_);
    cast(proj_w,   w_proj,  (size_t)HS_ * ED_);
    cast(attn_wqkv,w_qkv,   (size_t)L_ * 3 * HS_ * HS_);
    cast(attn_wo,  w_wo,    (size_t)L_ * HS_ * HS_);
    cast(spec_rw,  w_specr, (size_t)L_ * NE_ * HS_);
    cast(spec_f1w, w_spec1, (size_t)L_ * NE_ * FF_ * HS_);
    cast(spec_f2w, w_spec2, (size_t)L_ * NE_ * HS_ * FF_);
    cast(shr_rw,   w_shrr,  (size_t)L_ * NS_ * HS_);
    cast(shr_f1w,  w_shr1,  (size_t)L_ * NS_ * FF_ * HS_);
    cast(shr_f2w,  w_shr2,  (size_t)L_ * NS_ * HS_ * FF_);
    cast(cls_w,    w_cls,   (size_t)NC_ * HS_);

    auto gemm = [&](const bf16* A, const bf16* Bw, const float* bias,
                    float* Cf, bf16* Cb, const float* rs, int rss,
                    int M, int N, int K, int act, int accum) {
        dim3 g((N + 127) / 128, (M + 127) / 128);
        gemm_bf16_kernel<<<g, 256, 0, stream>>>(A, Bw, bias, Cf, Cb, rs, rss,
                                                M, N, K, act, accum);
    };

    // ---- tokenizer ----
    {
        const size_t total = T * 320;
        patchify_kernel<<<(int)((total + 255) / 256), 256, 0, stream>>>(x, t_bf);
    }
    gemm(t_bf, w_tok1, tok_b1, nullptr, z1_bf, nullptr, 0, T_TOK, SF_, 320, 1, 0);
    gemm(z1_bf, w_tok2, tok_b2, z2_f, nullptr, nullptr, 0, T_TOK, ED_, SF_, 1, 0);
    peadd_kernel<<<(T_TOK * ED_ + 255) / 256, 256, 0, stream>>>(z2_f, z2_bf);
    gemm(z2_bf, w_proj, proj_b, h_f, h_bf, nullptr, 0, T_TOK, HS_, ED_, 0, 0);

    // ---- transformer layers ----
    for (int l = 0; l < L_; ++l) {
        gemm(h_bf, w_qkv + (size_t)l * 3 * HS_ * HS_, attn_bqkv + (size_t)l * 3 * HS_,
             nullptr, qkv_bf, nullptr, 0, T_TOK, 3 * HS_, HS_, 0, 0);
        attn_kernel<<<dim3(S_TOK / 128, NH_, B_), 256, 0, stream>>>(qkv_bf, ao_bf);
        gemm(ao_bf, w_wo + (size_t)l * HS_ * HS_, attn_bo + (size_t)l * HS_,
             tmp_f, nullptr, nullptr, 0, T_TOK, HS_, HS_, 0, 0);
        ln_kernel<<<T_TOK, 128, 0, stream>>>(h_f, tmp_f,
             ln1_g + (size_t)l * HS_, ln1_b + (size_t)l * HS_, h_f, h_bf);

        // routers
        gemm(h_bf, w_specr + (size_t)l * NE_ * HS_, nullptr, lg1_f, nullptr,
             nullptr, 0, T_TOK, NE_, HS_, 0, 0);
        spec_router_kernel<<<(T_TOK + 255) / 256, 256, 0, stream>>>(lg1_f, comb_f);
        gemm(h_bf, w_shrr + (size_t)l * NS_ * HS_, nullptr, lg2_f, nullptr,
             nullptr, 0, T_TOK, NS_, HS_, 0, 0);
        shr_router_kernel<<<(T_TOK + 255) / 256, 256, 0, stream>>>(lg2_f, sp_f);

        // specific experts (dense, combine-weighted; first expert initializes acc)
        for (int e = 0; e < NE_; ++e) {
            const size_t wi = (size_t)l * NE_ + e;
            gemm(h_bf, w_spec1 + wi * FF_ * HS_, spec_f1b + wi * FF_,
                 nullptr, hid_bf, nullptr, 0, T_TOK, FF_, HS_, 2, 0);
            gemm(hid_bf, w_spec2 + wi * HS_ * FF_, spec_f2b + wi * HS_,
                 moe_f, nullptr, comb_f + e, NE_, T_TOK, HS_, FF_, 0, (e > 0) ? 1 : 0);
        }
        // shared experts (softmax-weighted, accumulate)
        for (int e = 0; e < NS_; ++e) {
            const size_t wi = (size_t)l * NS_ + e;
            gemm(h_bf, w_shr1 + wi * FF_ * HS_, shr_f1b + wi * FF_,
                 nullptr, hid_bf, nullptr, 0, T_TOK, FF_, HS_, 2, 0);
            gemm(hid_bf, w_shr2 + wi * HS_ * FF_, shr_f2b + wi * HS_,
                 moe_f, nullptr, sp_f + e, NS_, T_TOK, HS_, FF_, 0, 1);
        }

        ln_kernel<<<T_TOK, 128, 0, stream>>>(h_f, moe_f,
             lnm_g + (size_t)l * HS_, lnm_b + (size_t)l * HS_, m_f, nullptr);
        ln_kernel<<<T_TOK, 128, 0, stream>>>(h_f, m_f,
             ln2_g + (size_t)l * HS_, ln2_b + (size_t)l * HS_, h_f, h_bf);
    }

    // ---- head ----
    pool_kernel<<<(B_ * HS_ + 255) / 256, 256, 0, stream>>>(h_f, pool_bf);
    gemm(pool_bf, w_cls, cls_b, (float*)d_out, nullptr, nullptr, 0,
         B_, NC_, HS_, 0, 0);
}
